// SampledSoftmaxLoss_53884659696243
// MI455X (gfx1250) — matrix operand order repaired and checked by reading, not compiled, not run on hardware
//
#include <hip/hip_runtime.h>
#include <hip/hip_bf16.h>

// Problem constants (match reference).
#define KB 8192   // batch
#define KD 256    // embedding dim

typedef __attribute__((ext_vector_type(16))) __bf16 v16bf;
typedef __attribute__((ext_vector_type(8)))  __bf16 v8bf;
typedef __attribute__((ext_vector_type(8)))  float  v8f;

// ---------------------------------------------------------------------------
// Kernel 1: split f32 -> (hi, lo) bf16 pairs for both matrices.
// ---------------------------------------------------------------------------
__global__ void ssl_convert_kernel(const float* __restrict__ inp,
                                   const float* __restrict__ tgt,
                                   __bf16* __restrict__ Ahi, __bf16* __restrict__ Alo,
                                   __bf16* __restrict__ Bhi, __bf16* __restrict__ Blo) {
    int idx = blockIdx.x * blockDim.x + threadIdx.x;   // one element of each matrix
    float x = inp[idx];
    __bf16 h = (__bf16)x;
    Ahi[idx] = h;
    Alo[idx] = (__bf16)(x - (float)h);
    float y = tgt[idx];
    __bf16 g = (__bf16)y;
    Bhi[idx] = g;
    Blo[idx] = (__bf16)(y - (float)g);
}

// ---------------------------------------------------------------------------
// Kernel 2: per-row n_miss count, row correction log1p(-q_i)-log(n_miss_i),
//           per-col log(q_j), and zero the S accumulator (ws is poisoned).
// ---------------------------------------------------------------------------
__global__ void ssl_rows_kernel(const int* __restrict__ ids,
                                const float* __restrict__ q,
                                float* __restrict__ rowc,
                                float* __restrict__ lq,
                                float* __restrict__ S) {
    __shared__ int cnt[256];
    const int i   = blockIdx.x;
    const int tid = threadIdx.x;
    const int myid = ids[i];
    int c = 0;
    for (int j = tid; j < KB; j += 256) c += (ids[j] == myid) ? 1 : 0;
    cnt[tid] = c;
    __syncthreads();
    for (int off = 128; off > 0; off >>= 1) {
        if (tid < off) cnt[tid] += cnt[tid + off];
        __syncthreads();
    }
    if (tid == 0) {
        int n_miss = KB - cnt[0];                 // id_i==id_i removes the diagonal
        rowc[i] = log1pf(-q[i]) - logf((float)n_miss);
        lq[i]   = logf(q[i]);
        S[i]    = 0.0f;
    }
}

// ---------------------------------------------------------------------------
// Kernel 3: exact f32 diagonal pos_sim[i] = <input[i], target[i]>.
// One wave per row; 8 waves per block.
// ---------------------------------------------------------------------------
__global__ void ssl_diag_kernel(const float* __restrict__ A,
                                const float* __restrict__ Bm,
                                float* __restrict__ pos) {
    const int w    = threadIdx.x >> 5;
    const int lane = threadIdx.x & 31;
    const int i    = blockIdx.x * 8 + w;
    const float* a = A  + (size_t)i * KD;
    const float* b = Bm + (size_t)i * KD;
    float s = 0.0f;
    #pragma unroll
    for (int k = 0; k < KD / 32; ++k) s = fmaf(a[lane + 32 * k], b[lane + 32 * k], s);
    #pragma unroll
    for (int m = 16; m >= 1; m >>= 1) s += __shfl_xor(s, m, 32);
    if (lane == 0) pos[i] = s;
}

// ---------------------------------------------------------------------------
// WMMA fragment loaders (layouts per CDNA5 ISA 7.12.2, wave32).
// A (16x32 bf16, MxK): lane L -> M = L&15; half = L>>4;
//   vector elems 0..7  -> K = k0 + half*8 + e        (16B contiguous)
//   vector elems 8..15 -> K = k0 + 16 + half*8 + e-8 (16B contiguous)
// B (32x16 bf16, KxN): lane L -> N = L&15; half = L>>4;
//   vector elems 0..7  -> K = k0 + half*8 + e        (16B contiguous)
//   vector elems 8..15 -> K = k0 + 16 + half*8 + e-8 (16B contiguous)
// Both source matrices are [KB x KD] row-major; for B we read target rows
// (j) along K, which is exactly the transposed operand sim needs.
// ---------------------------------------------------------------------------
__device__ __forceinline__ v16bf ssl_load_a(const __bf16* __restrict__ M,
                                            int row, int k0, int half) {
    const __bf16* base = M + (size_t)row * KD + k0 + half * 8;
    union { v16bf v; v8bf h[2]; } u;
    u.h[0] = *reinterpret_cast<const v8bf*>(base);
    u.h[1] = *reinterpret_cast<const v8bf*>(base + 16);
    return u.v;
}

__device__ __forceinline__ v16bf ssl_load_b(const __bf16* __restrict__ M,
                                            int row, int k0, int half) {
    const __bf16* base = M + (size_t)row * KD + k0 + half * 8;
    union { v16bf v; v8bf h[2]; } u;
    u.h[0] = *reinterpret_cast<const v8bf*>(base);
    u.h[1] = *reinterpret_cast<const v8bf*>(base + 16);
    return u.v;
}

__device__ __forceinline__ v8f ssl_wmma(v16bf a, v16bf b, v8f c) {
    return __builtin_amdgcn_wmma_f32_16x16x32_bf16(
        /*neg_a=*/false, a, /*neg_b=*/false, b,
        /*c_mod=*/(short)0, c, /*reuse_a=*/false, /*reuse_b=*/false);
}

// ---------------------------------------------------------------------------
// Kernel 4: fused GEMM + masked logQ-corrected exp + row reduction.
// Block = 256 threads = 8 waves; wave tile 64x64 (4x4 WMMA frags, f32 acc);
// WG covers 256 (M) x 128 (N). Split-bf16: sim ~= ah*bh + ah*bl + al*bh.
// ---------------------------------------------------------------------------
__global__ void __launch_bounds__(256)
ssl_gemm_kernel(const __bf16* __restrict__ Ahi, const __bf16* __restrict__ Alo,
                const __bf16* __restrict__ Bhi, const __bf16* __restrict__ Blo,
                const int*    __restrict__ ids,
                const float*  __restrict__ rowc,
                const float*  __restrict__ lq,
                float*        __restrict__ S) {
    const int wave  = threadIdx.x >> 5;
    const int lane  = threadIdx.x & 31;
    const int cidx  = lane & 15;
    const int half  = lane >> 4;
    const int waveM = wave & 3;        // 4 waves along M
    const int waveN = wave >> 2;       // 2 waves along N
    const int RM = blockIdx.y * 256 + waveM * 64;
    const int CN = blockIdx.x * 128 + waveN * 64;

    v8f acc[4][4];
    const v8f vzero = {};
    #pragma unroll
    for (int tm = 0; tm < 4; ++tm)
        #pragma unroll
        for (int tn = 0; tn < 4; ++tn) acc[tm][tn] = vzero;

    for (int k0 = 0; k0 < KD; k0 += 32) {
        v16bf ah[4], al[4], bh[4], bl[4];
        #pragma unroll
        for (int tm = 0; tm < 4; ++tm) {
            const int row = RM + tm * 16 + cidx;
            ah[tm] = ssl_load_a(Ahi, row, k0, half);
            al[tm] = ssl_load_a(Alo, row, k0, half);
        }
        #pragma unroll
        for (int tn = 0; tn < 4; ++tn) {
            const int col = CN + tn * 16 + cidx;
            bh[tn] = ssl_load_b(Bhi, col, k0, half);
            bl[tn] = ssl_load_b(Blo, col, k0, half);
        }
        #pragma unroll
        for (int tm = 0; tm < 4; ++tm)
            #pragma unroll
            for (int tn = 0; tn < 4; ++tn) {
                acc[tm][tn] = ssl_wmma(al[tm], bh[tn], acc[tm][tn]);
                acc[tm][tn] = ssl_wmma(ah[tm], bl[tn], acc[tm][tn]);
                acc[tm][tn] = ssl_wmma(ah[tm], bh[tn], acc[tm][tn]);
            }
    }

    // Epilogue: column-side values once per wave.
    int   idj[4];
    float lqj[4];
    #pragma unroll
    for (int tn = 0; tn < 4; ++tn) {
        const int j = CN + tn * 16 + cidx;
        idj[tn] = ids[j];
        lqj[tn] = lq[j];
    }

    // C layout: elem r of lane L -> row M = r + half*8 (same for all cidx),
    // col N = cidx. Reduce over the 16 lanes of each half, then one atomic.
    #pragma unroll
    for (int tm = 0; tm < 4; ++tm) {
        #pragma unroll
        for (int r = 0; r < 8; ++r) {
            const int i  = RM + tm * 16 + half * 8 + r;
            const int idi = ids[i];
            const float rc = rowc[i];
            float part = 0.0f;
            #pragma unroll
            for (int tn = 0; tn < 4; ++tn) {
                const float s = acc[tm][tn][r];
                if (idi != idj[tn]) part += __expf(s + rc - lqj[tn]);
            }
            #pragma unroll
            for (int m = 8; m >= 1; m >>= 1) part += __shfl_xor(part, m, 32);
            if (cidx == 0) atomicAdd(&S[i], part);
        }
    }
}

// ---------------------------------------------------------------------------
// Kernel 5: loss = mean_i( -pos_i + log(exp(pos_i) + S_i) ).
// ---------------------------------------------------------------------------
__global__ void ssl_final_kernel(const float* __restrict__ pos,
                                 const float* __restrict__ S,
                                 float* __restrict__ out) {
    __shared__ float red[256];
    const int tid = threadIdx.x;
    float local = 0.0f;
    for (int i = tid; i < KB; i += 256) {
        const float p = pos[i];
        local += -p + logf(expf(p) + S[i]);
    }
    red[tid] = local;
    __syncthreads();
    for (int off = 128; off > 0; off >>= 1) {
        if (tid < off) red[tid] += red[tid + off];
        __syncthreads();
    }
    if (tid == 0) out[0] = red[0] / (float)KB;
}

// ---------------------------------------------------------------------------
// Launch
// ---------------------------------------------------------------------------
extern "C" void kernel_launch(void* const* d_in, const int* in_sizes, int n_in,
                              void* d_out, int out_size, void* d_ws, size_t ws_size,
                              hipStream_t stream) {
    const float* input_emb  = (const float*)d_in[0];
    const float* target_emb = (const float*)d_in[1];
    const int*   target_ids = (const int*)d_in[2];
    const float* q_probas   = (const float*)d_in[3];
    float* out = (float*)d_out;

    // Workspace partition (all chunks naturally 256B-aligned).
    const size_t nmat = (size_t)KB * KD;          // 2M elements
    char* w = (char*)d_ws;
    __bf16* Ahi = (__bf16*)w; w += nmat * sizeof(__bf16);
    __bf16* Alo = (__bf16*)w; w += nmat * sizeof(__bf16);
    __bf16* Bhi = (__bf16*)w; w += nmat * sizeof(__bf16);
    __bf16* Blo = (__bf16*)w; w += nmat * sizeof(__bf16);
    float* S    = (float*)w;  w += KB * sizeof(float);
    float* pos  = (float*)w;  w += KB * sizeof(float);
    float* rowc = (float*)w;  w += KB * sizeof(float);
    float* lq   = (float*)w;  w += KB * sizeof(float);

    ssl_convert_kernel<<<dim3(nmat / 256), dim3(256), 0, stream>>>(
        input_emb, target_emb, Ahi, Alo, Bhi, Blo);

    ssl_rows_kernel<<<dim3(KB), dim3(256), 0, stream>>>(
        target_ids, q_probas, rowc, lq, S);

    ssl_diag_kernel<<<dim3(KB / 8), dim3(256), 0, stream>>>(
        input_emb, target_emb, pos);

    ssl_gemm_kernel<<<dim3(KB / 128, KB / 256), dim3(256), 0, stream>>>(
        Ahi, Alo, Bhi, Blo, target_ids, rowc, lq, S);

    ssl_final_kernel<<<dim3(1), dim3(256), 0, stream>>>(pos, S, out);
}
